// CQAttentionLayer_74766790689288
// MI455X (gfx1250) — compile-verified, matrix operand order checked by reading, not run on hardware
//
#include <hip/hip_runtime.h>

// CQAttention for MI455X (gfx1250, wave32, WMMA + TDM).
// b = s1 @ (s2^T @ ctx) avoids the CLxCL intermediate.
// GEMMs: v_wmma_f32_16x16x32_bf16, f32 accumulate.
// Softmax statistics fused into the scores epilogue (saves ~100MB HBM traffic).

#define B_  16
#define H_  128
#define CL_ 2048
#define QL_ 256

typedef __attribute__((ext_vector_type(16))) __bf16 v16bf;
typedef __attribute__((ext_vector_type(8)))  float  v8f;
typedef unsigned int v4ui __attribute__((ext_vector_type(4)));
typedef int v8i_t __attribute__((ext_vector_type(8)));
typedef int v4i_t __attribute__((ext_vector_type(4)));

#if defined(__AMDGCN__) && __has_builtin(__builtin_amdgcn_tensor_load_to_lds) && \
    __has_builtin(__builtin_amdgcn_s_wait_tensorcnt)
#define USE_TDM 1
#endif

#ifdef USE_TDM
// Issue one TDM 2D tile load (bf16 elements) global -> LDS, with optional LDS
// row padding. Descriptor layout per CDNA5 ISA section 8 (groups 0/1; groups
// 2/3 zero => dims 2+ unused).
__device__ __forceinline__ void tdm_load_tile_bf16(unsigned lds_byte_off,
                                                   const void* gptr,
                                                   unsigned rows, unsigned cols,
                                                   unsigned row_stride_elems,
                                                   unsigned pad_interval_code,
                                                   unsigned pad_amount_code,
                                                   bool pad_en) {
  const unsigned long long ga = (unsigned long long)gptr;
  v4ui g0;
  g0[0] = 1u;                                            // count=1, user mode
  g0[1] = lds_byte_off;                                  // lds_addr
  g0[2] = (unsigned)ga;                                  // global_addr[31:0]
  g0[3] = (unsigned)((ga >> 32) & 0x01FFFFFFu) | (2u << 30);  // addr[56:32] | type=2
  v8i_t g1;
  g1[0] = (int)((1u << 16)                               // data_size = 2 bytes
                | (pad_en ? (1u << 20) : 0u)             // pad_enable
                | (pad_interval_code << 22)
                | (pad_amount_code << 25));
  g1[1] = (int)((cols & 0xFFFFu) << 16);                 // tensor_dim0[15:0] @bit48
  g1[2] = (int)(((cols >> 16) & 0xFFFFu) | ((rows & 0xFFFFu) << 16));  // dim0 hi | dim1 lo
  g1[3] = (int)(((rows >> 16) & 0xFFFFu) | ((cols & 0xFFFFu) << 16));  // dim1 hi | tile_dim0
  g1[4] = (int)(rows & 0xFFFFu);                         // tile_dim1 | tile_dim2=0
  g1[5] = (int)row_stride_elems;                         // tensor_dim0_stride lo
  g1[6] = 0;                                             // stride hi | dim1_stride lo
  g1[7] = 0;
  v4i_t z = {0, 0, 0, 0};
#if __has_include(<hip/amd_detail/amd_gfx1250_TDM.h>)
  v8i_t z8 = {0, 0, 0, 0, 0, 0, 0, 0};
  __builtin_amdgcn_tensor_load_to_lds(g0, g1, z, z, z8, 0);
#else
  __builtin_amdgcn_tensor_load_to_lds(g0, g1, z, z, 0);
#endif
}
#endif

// Per-lane K mapping for 16-bit A/B fragments (CDNA5 ISA 7.12.2):
// lane<16: K = {0..7, 16..23}; lane>=16: K = {8..15, 24..31} (relative to kb).
__device__ __forceinline__ v16bf frag_load(const __bf16* row, int kb, int khalf) {
  v16bf f;
#pragma unroll
  for (int i = 0; i < 16; ++i) {
    const int k = (i < 8) ? (kb + khalf + i) : (kb + 8 + khalf + i);
    f[i] = row[k];
  }
  return f;
}

// ---------------------------------------------------------------- prep
__global__ __launch_bounds__(256) void k_prep(const float* __restrict__ ctx,
                                              const float* __restrict__ qry,
                                              const float* __restrict__ w,
                                              float* __restrict__ cw,
                                              float* __restrict__ qw,
                                              float* __restrict__ rowsum,
                                              float* __restrict__ colsum) {
  const int idx = blockIdx.x * 256 + threadIdx.x;
  const int N1 = B_ * CL_;
  if (idx < N1) {
    const int b = idx >> 11, c = idx & (CL_ - 1);
    float s = 0.f;
    for (int h = 0; h < H_; ++h)
      s += ctx[((size_t)(b * H_ + h)) * CL_ + c] * w[H_ + h];
    cw[idx] = s;
    rowsum[idx] = 0.f;
  } else {
    const int j = idx - N1;
    if (j < B_ * QL_) {
      const int b = j >> 8, q = j & (QL_ - 1);
      float s = 0.f;
      for (int h = 0; h < H_; ++h)
        s += qry[((size_t)(b * H_ + h)) * QL_ + q] * w[h];
      qw[j] = s;
      colsum[j] = 0.f;
    }
  }
}

// ---------------------------------------------------------------- scores (+ fused exp & masked row/col sums)
// Block: 32c x 128q tile, K=H=128. Writes e=exp(s) to S; atomics for sums.
#define SC_AST 136
#define SC_BST 136
#define SC_EST 132
__global__ __launch_bounds__(256) void k_scores(const float* __restrict__ ctx,
                                                const float* __restrict__ qry,
                                                const float* __restrict__ w,
                                                const float* __restrict__ cw,
                                                const float* __restrict__ qw,
                                                const int* __restrict__ cmask,
                                                const int* __restrict__ qmask,
                                                float* __restrict__ S,
                                                float* __restrict__ rowsum,
                                                float* __restrict__ colsum) {
  __shared__ __bf16 At[32 * SC_AST];   // [c][k]
  __shared__ __bf16 Bt[128 * SC_BST];  // [q][k]
  __shared__ float  E[32 * SC_EST];    // e tile for reductions
  const int tid = threadIdx.x;
  const int b  = blockIdx.x >> 6;
  const int c0 = (blockIdx.x & 63) << 5;
  const int q0 = blockIdx.y << 7;

  {  // A tile: transpose-load + scale by wcq + cvt
    const int c = tid & 31, kg = tid >> 5;
#pragma unroll
    for (int kk = 0; kk < 16; ++kk) {
      const int k = kk * 8 + kg;
      const float v = ctx[((size_t)(b * H_ + k)) * CL_ + c0 + c] * w[2 * H_ + k];
      At[c * SC_AST + k] = (__bf16)v;
    }
  }
  {  // B tile: [q][k]
    const int q = tid & 127, kg = tid >> 7;
    for (int kk = 0; kk < 64; ++kk) {
      const int k = kk * 2 + kg;
      Bt[q * SC_BST + k] = (__bf16)qry[((size_t)(b * H_ + k)) * QL_ + q0 + q];
    }
  }
  __syncthreads();

  const int wv = tid >> 5, lane = tid & 31;
  const int khalf = (lane >= 16) ? 8 : 0;
  const int qn = wv << 4;
  v8f acc0 = {}, acc1 = {};
  const __bf16* arow0 = &At[(lane & 15) * SC_AST];
  const __bf16* arow1 = &At[((lane & 15) + 16) * SC_AST];
  const __bf16* brow  = &Bt[(qn + (lane & 15)) * SC_BST];
#pragma unroll
  for (int kb = 0; kb < 128; kb += 32) {
    v16bf bf = frag_load(brow, kb, khalf);
    v16bf a0 = frag_load(arow0, kb, khalf);
    acc0 = __builtin_amdgcn_wmma_f32_16x16x32_bf16(false, a0, false, bf, (short)0, acc0, false, false);
    v16bf a1 = frag_load(arow1, kb, khalf);
    acc1 = __builtin_amdgcn_wmma_f32_16x16x32_bf16(false, a1, false, bf, (short)0, acc1, false, false);
  }
  const int n = lane & 15;
  const int ql = qn + n;
  const int q = q0 + ql;
  const float qwv = qw[b * QL_ + q];
#pragma unroll
  for (int r = 0; r < 8; ++r) {
    const int cl = r + khalf;
    const int c = c0 + cl;
    const float e = __expf(acc0[r] + cw[b * CL_ + c] + qwv);
    S[((size_t)b * CL_ + c) * QL_ + q] = e;
    E[cl * SC_EST + ql] = e;
  }
#pragma unroll
  for (int r = 0; r < 8; ++r) {
    const int cl = 16 + r + khalf;
    const int c = c0 + cl;
    const float e = __expf(acc1[r] + cw[b * CL_ + c] + qwv);
    S[((size_t)b * CL_ + c) * QL_ + q] = e;
    E[cl * SC_EST + ql] = e;
  }
  __syncthreads();
  if (tid < 32) {  // masked row partial sums (softmax over q)
    float s = 0.f;
    for (int qq = 0; qq < 128; ++qq)
      s += (qmask[b * QL_ + q0 + qq] > 0) ? E[tid * SC_EST + qq] : 0.f;
    atomicAdd(&rowsum[b * CL_ + c0 + tid], s);
  } else if (tid < 160) {  // masked col partial sums (softmax over c)
    const int qq = tid - 32;
    float s = 0.f;
    for (int cc = 0; cc < 32; ++cc)
      s += (cmask[b * CL_ + c0 + cc] > 0) ? E[cc * SC_EST + qq] : 0.f;
    atomicAdd(&colsum[b * QL_ + q0 + qq], s);
  }
}

// ---------------------------------------------------------------- normalize -> s1,s2 (bf16)
__global__ __launch_bounds__(256) void k_normalize(const float* __restrict__ S,
                                                   const float* __restrict__ rowsum,
                                                   const float* __restrict__ colsum,
                                                   const int* __restrict__ cmask,
                                                   const int* __restrict__ qmask,
                                                   __bf16* __restrict__ s1,
                                                   __bf16* __restrict__ s2) {
  const size_t idx = (size_t)blockIdx.x * 256 + threadIdx.x;
  const int q = (int)(idx & (QL_ - 1));
  const int c = (int)((idx >> 8) & (CL_ - 1));
  const int b = (int)(idx >> 19);
  const float e = S[idx];
  const float rs = rowsum[(b << 11) + c];
  const float cs = colsum[(b << 8) + q];
  s1[idx] = (__bf16)((qmask[(b << 8) + q] > 0 && rs > 0.f) ? (e / rs) : 0.0f);
  s2[idx] = (__bf16)((cmask[(b << 11) + c] > 0 && cs > 0.f) ? (e / cs) : 0.0f);
}

// ---------------------------------------------------------------- t[b,h,q] = sum_c s2[b,c,q]*ctx[b,h,c]
#define T_AST 72
#define T_BST 72
__global__ __launch_bounds__(256) void k_t(const float* __restrict__ ctx,
                                           const __bf16* __restrict__ s2,
                                           __bf16* __restrict__ tbf) {
  __shared__ __bf16 At[128 * T_AST];  // [h][c]
  __shared__ __bf16 Bt[64 * T_BST];   // [q][c]
  const int tid = threadIdx.x;
  const int b  = blockIdx.x >> 2;
  const int q0 = (blockIdx.x & 3) << 6;
  const int wv = tid >> 5, lane = tid & 31;
  const int khalf = (lane >= 16) ? 8 : 0;
  const int hm = wv << 4;
  v8f acc[4] = {{}, {}, {}, {}};

  for (int c0 = 0; c0 < CL_; c0 += 64) {
    __syncthreads();
    {  // A: ctx chunk (K=c contiguous in memory), f32->bf16
      const int cc = tid & 63, hg = tid >> 6;
      if (c0 + 64 < CL_)
        __builtin_prefetch(&ctx[((size_t)(b * H_ + hg)) * CL_ + c0 + 64 + cc], 0, 3);
#pragma unroll
      for (int hh = 0; hh < 32; ++hh) {
        const int h = hh * 4 + hg;
        At[h * T_AST + cc] = (__bf16)ctx[((size_t)(b * H_ + h)) * CL_ + c0 + cc];
      }
    }
    {  // B: s2 tile transposed into [q][c]
      const int qq = tid & 63, cg = tid >> 6;
#pragma unroll
      for (int ci = 0; ci < 16; ++ci) {
        const int cl = ci * 4 + cg;
        Bt[qq * T_BST + cl] = s2[((size_t)b * CL_ + c0 + cl) * QL_ + q0 + qq];
      }
    }
    __syncthreads();
    const __bf16* arow = &At[(hm + (lane & 15)) * T_AST];
#pragma unroll
    for (int kb = 0; kb < 64; kb += 32) {
      v16bf af = frag_load(arow, kb, khalf);
#pragma unroll
      for (int j = 0; j < 4; ++j) {
        const __bf16* brow = &Bt[((j << 4) + (lane & 15)) * T_BST];
        v16bf bf = frag_load(brow, kb, khalf);
        acc[j] = __builtin_amdgcn_wmma_f32_16x16x32_bf16(false, af, false, bf, (short)0, acc[j], false, false);
      }
    }
  }
#pragma unroll
  for (int j = 0; j < 4; ++j) {
    const int q = q0 + (j << 4) + (lane & 15);
#pragma unroll
    for (int r = 0; r < 8; ++r) {
      const int h = hm + r + khalf;
      tbf[((size_t)(b * H_ + h)) * QL_ + q] = (__bf16)acc[j][r];
    }
  }
}

// ---------------------------------------------------------------- a = s1 @ qry ; out rows [H,2H) and [2H,3H)
#define AB_AST 264  // 256 + 8 pad => rows = 512B data + 16B pad (TDM: interval code 6, amount code 3)
#define AB_BST 264
__global__ __launch_bounds__(128) void k_a(const float* __restrict__ ctx,
                                           const float* __restrict__ qry,
                                           const __bf16* __restrict__ s1,
                                           float* __restrict__ out) {
  __shared__ __bf16 At[16 * AB_AST];  // s1 [c][q]
  __shared__ __bf16 Bt[64 * AB_BST];  // qry [h][q]
  __shared__ float Cst[16 * 65];
  const int tid = threadIdx.x;
  const int b  = blockIdx.x >> 7;
  const int c0 = (blockIdx.x & 127) << 4;
  const int h0 = blockIdx.y << 6;
#ifdef USE_TDM
  if (tid < 32)  // wave 0 issues the DMA for the bf16 A tile
    tdm_load_tile_bf16((unsigned)(unsigned long long)&At[0],
                       &s1[((size_t)b * CL_ + c0) * QL_], 16, 256, QL_, 6, 3, true);
#else
#pragma unroll
  for (int it = 0; it < 32; ++it) {
    const int e = it * 128 + tid;
    At[(e >> 8) * AB_AST + (e & 255)] =
        s1[((size_t)b * CL_ + c0 + (e >> 8)) * QL_ + (e & 255)];
  }
#endif
  for (int it = 0; it < 128; ++it) {  // f32 -> bf16 conversion: stays on VALU path
    const int e = it * 128 + tid;
    Bt[(e >> 8) * AB_BST + (e & 255)] =
        (__bf16)qry[((size_t)(b * H_ + h0 + (e >> 8))) * QL_ + (e & 255)];
  }
#ifdef USE_TDM
  __builtin_amdgcn_s_wait_tensorcnt(0);
#endif
  __syncthreads();
  const int wv = tid >> 5, lane = tid & 31;
  const int khalf = (lane >= 16) ? 8 : 0;
  v8f acc = {};
  const __bf16* arow = &At[(lane & 15) * AB_AST];
  const __bf16* brow = &Bt[((wv << 4) + (lane & 15)) * AB_BST];
#pragma unroll
  for (int kb = 0; kb < 256; kb += 32) {
    v16bf af = frag_load(arow, kb, khalf);
    v16bf bf = frag_load(brow, kb, khalf);
    acc = __builtin_amdgcn_wmma_f32_16x16x32_bf16(false, af, false, bf, (short)0, acc, false, false);
  }
  {
    const int n = (wv << 4) + (lane & 15);
#pragma unroll
    for (int r = 0; r < 8; ++r) Cst[(r + khalf) * 65 + n] = acc[r];
  }
  __syncthreads();
#pragma unroll
  for (int it = 0; it < 8; ++it) {
    const int e = it * 128 + tid;
    const int hl = e >> 4, cl = e & 15;
    const float av = Cst[cl * 65 + hl];
    const float cv = ctx[((size_t)(b * H_ + h0 + hl)) * CL_ + c0 + cl];
    const size_t ob = (size_t)b * 4 * H_ * CL_;
    out[ob + (size_t)(H_ + h0 + hl) * CL_ + c0 + cl] = av;
    out[ob + (size_t)(2 * H_ + h0 + hl) * CL_ + c0 + cl] = av * cv;
  }
}

// ---------------------------------------------------------------- bmat = s1 @ t ; out rows [3H,4H) = ctx*b
__global__ __launch_bounds__(128) void k_b(const float* __restrict__ ctx,
                                           const __bf16* __restrict__ tbf,
                                           const __bf16* __restrict__ s1,
                                           float* __restrict__ out) {
  __shared__ __bf16 At[16 * AB_AST];  // s1 [c][q]
  __shared__ __bf16 Bt[64 * AB_BST];  // t  [h][q]
  __shared__ float Cst[16 * 65];
  const int tid = threadIdx.x;
  const int b  = blockIdx.x >> 7;
  const int c0 = (blockIdx.x & 127) << 4;
  const int h0 = blockIdx.y << 6;
#ifdef USE_TDM
  if (tid < 32) {  // wave 0 issues both bf16 tile DMAs
    tdm_load_tile_bf16((unsigned)(unsigned long long)&At[0],
                       &s1[((size_t)b * CL_ + c0) * QL_], 16, 256, QL_, 6, 3, true);
    tdm_load_tile_bf16((unsigned)(unsigned long long)&Bt[0],
                       &tbf[((size_t)(b * H_ + h0)) * QL_], 64, 256, QL_, 6, 3, true);
  }
  __builtin_amdgcn_s_wait_tensorcnt(0);
#else
#pragma unroll
  for (int it = 0; it < 32; ++it) {
    const int e = it * 128 + tid;
    At[(e >> 8) * AB_AST + (e & 255)] =
        s1[((size_t)b * CL_ + c0 + (e >> 8)) * QL_ + (e & 255)];
  }
  for (int it = 0; it < 128; ++it) {
    const int e = it * 128 + tid;
    Bt[(e >> 8) * AB_BST + (e & 255)] =
        tbf[((size_t)(b * H_ + h0 + (e >> 8))) * QL_ + (e & 255)];
  }
#endif
  __syncthreads();
  const int wv = tid >> 5, lane = tid & 31;
  const int khalf = (lane >= 16) ? 8 : 0;
  v8f acc = {};
  const __bf16* arow = &At[(lane & 15) * AB_AST];
  const __bf16* brow = &Bt[((wv << 4) + (lane & 15)) * AB_BST];
#pragma unroll
  for (int kb = 0; kb < 256; kb += 32) {
    v16bf af = frag_load(arow, kb, khalf);
    v16bf bf = frag_load(brow, kb, khalf);
    acc = __builtin_amdgcn_wmma_f32_16x16x32_bf16(false, af, false, bf, (short)0, acc, false, false);
  }
  {
    const int n = (wv << 4) + (lane & 15);
#pragma unroll
    for (int r = 0; r < 8; ++r) Cst[(r + khalf) * 65 + n] = acc[r];
  }
  __syncthreads();
#pragma unroll
  for (int it = 0; it < 8; ++it) {
    const int e = it * 128 + tid;
    const int hl = e >> 4, cl = e & 15;
    const float bv = Cst[cl * 65 + hl];
    const float cv = ctx[((size_t)(b * H_ + h0 + hl)) * CL_ + c0 + cl];
    out[(size_t)b * 4 * H_ * CL_ + (size_t)(3 * H_ + h0 + hl) * CL_ + c0 + cl] = bv * cv;
  }
}

// ---------------------------------------------------------------- out rows [0,H) = ctx (float4 copy)
__global__ __launch_bounds__(256) void k_copyctx(const float* __restrict__ ctx,
                                                 float* __restrict__ out) {
  const size_t i4 = (size_t)blockIdx.x * 256 + threadIdx.x;
  const int b = (int)(i4 >> 16);
  const size_t rem = i4 & 65535;
  const float4 v = ((const float4*)ctx)[i4];
  ((float4*)out)[(size_t)b * (4 * H_ * CL_ / 4) + rem] = v;
}

// ---------------------------------------------------------------- launch
extern "C" void kernel_launch(void* const* d_in, const int* in_sizes, int n_in,
                              void* d_out, int out_size, void* d_ws, size_t ws_size,
                              hipStream_t stream) {
  const float* ctx   = (const float*)d_in[0];  // (B,H,CL)
  const float* qry   = (const float*)d_in[1];  // (B,H,QL)
  const int*   cmask = (const int*)d_in[2];    // (B,CL)
  const int*   qmask = (const int*)d_in[3];    // (B,QL)
  const float* w     = (const float*)d_in[4];  // (3H)
  float* out = (float*)d_out;                  // (B,4H,CL)

  char* p = (char*)d_ws;  // ~65.3 MB
  float*  S      = (float*)p;  p += (size_t)B_ * CL_ * QL_ * 4;
  __bf16* s1     = (__bf16*)p; p += (size_t)B_ * CL_ * QL_ * 2;
  __bf16* s2     = (__bf16*)p; p += (size_t)B_ * CL_ * QL_ * 2;
  __bf16* tbf    = (__bf16*)p; p += (size_t)B_ * H_ * QL_ * 2;
  float*  cw     = (float*)p;  p += (size_t)B_ * CL_ * 4;
  float*  qw     = (float*)p;  p += (size_t)B_ * QL_ * 4;
  float*  rowsum = (float*)p;  p += (size_t)B_ * CL_ * 4;
  float*  colsum = (float*)p;  p += (size_t)B_ * QL_ * 4;

  k_prep<<<(B_ * CL_ + B_ * QL_) / 256, 256, 0, stream>>>(ctx, qry, w, cw, qw, rowsum, colsum);
  k_scores<<<dim3(B_ * (CL_ / 32), QL_ / 128), 256, 0, stream>>>(
      ctx, qry, w, cw, qw, cmask, qmask, S, rowsum, colsum);
  k_normalize<<<(B_ * CL_ * QL_) / 256, 256, 0, stream>>>(S, rowsum, colsum, cmask, qmask, s1, s2);
  k_t<<<B_ * (QL_ / 64), 256, 0, stream>>>(ctx, s2, tbf);
  k_a<<<dim3(B_ * (CL_ / 16), H_ / 64), 128, 0, stream>>>(ctx, qry, s1, out);
  k_b<<<dim3(B_ * (CL_ / 16), H_ / 64), 128, 0, stream>>>(ctx, tbf, s1, out);
  k_copyctx<<<(B_ * H_ * CL_ / 4) / 256, 256, 0, stream>>>(ctx, out);
}
